// GCN_5995774345967
// MI455X (gfx1250) — compile-verified, matrix operand order checked by reading, not run on hardware
//
#include <hip/hip_runtime.h>
#include <hip/hip_bf16.h>

typedef __attribute__((ext_vector_type(16))) __bf16 v16bf;
typedef __attribute__((ext_vector_type(8)))  float  v8f;

#define N_NODES  100000
#define N_EDGES  640000
#define D_FEAT   128
#define HIDDEN   128
#define FINAL_D  64
#define N_GRAPHS 128
#define N_TILES32 (N_NODES / 32)   // 3125, exact

// ---- workspace layout (bytes) ----
#define OFF_AGG   ((size_t)0)                       // 100000*128*4 = 51,200,000
#define OFF_DEG   ((size_t)51200000)                // 100000*4
#define OFF_PMAX  ((size_t)51600000)                // 128*128*4 (int bits)
#define OFF_PSUM  ((size_t)51665536)                // 128*128*4
#define OFF_CNT   ((size_t)51731072)                // 128*4 (int)
#define OFF_ZBF   ((size_t)51731584)                // 128*256*2 bf16
#define ZERO_BYTES ((size_t)51731584)               // agg..cnt inclusive

union FragU { uint4 q[2]; v16bf v; };

__device__ __forceinline__ v8f wmma_bf16(v16bf a, v16bf b, v8f c) {
    return __builtin_amdgcn_wmma_f32_16x16x32_bf16(
        /*neg_a=*/false, a, /*neg_b=*/false, b,
        /*c_mod=*/(short)0, c, /*reuse_a=*/false, /*reuse_b=*/false);
}

// A-fragment (16x32 bf16) from fp32 row-major row pointer; this lane's K chunks
// are [k0..k0+7] and [k0+16..k0+23], k0 = ktile*32 + (lane>>4)*8.
__device__ __forceinline__ v16bf make_a_frag_f32(const float* __restrict__ rowp,
                                                 int k0, float s) {
    const float4* p0 = (const float4*)(rowp + k0);
    const float4* p1 = (const float4*)(rowp + k0 + 16);
    float4 a = p0[0], b = p0[1], c = p1[0], d = p1[1];
    float f[16] = {a.x,a.y,a.z,a.w, b.x,b.y,b.z,b.w,
                   c.x,c.y,c.z,c.w, d.x,d.y,d.z,d.w};
    v16bf v;
#pragma unroll
    for (int i = 0; i < 16; ++i) v[i] = (__bf16)(f[i] * s);
    return v;
}

// Fragment from a row-major bf16 row pointer (LDS or global); same per-lane
// K pattern (two 16-byte chunks). Used for B-fragments (n = lane%16 selects row
// of W[n][k], i.e. column of B) and for A-fragments of the head GEMM.
__device__ __forceinline__ v16bf make_frag_bf16(const __bf16* __restrict__ rowp, int k0) {
    FragU u;
    u.q[0] = *(const uint4*)(rowp + k0);
    u.q[1] = *(const uint4*)(rowp + k0 + 16);
    return u.v;
}

// ---------------- kernel 0: zero workspace ----------------
__global__ __launch_bounds__(256) void init_ws(uint4* ws, size_t n16) {
    size_t i = (size_t)blockIdx.x * 256 + threadIdx.x;
    uint4 z = {0u, 0u, 0u, 0u};
    for (; i < n16; i += (size_t)gridDim.x * 256) ws[i] = z;
}

// ---------------- kernel 1: edge scatter (mean numerator + degree) ----------------
__global__ __launch_bounds__(256) void scatter_edges(const float* __restrict__ x,
                                                     const int* __restrict__ src,
                                                     const int* __restrict__ dst,
                                                     float* __restrict__ agg,
                                                     float* __restrict__ deg) {
    int t = blockIdx.x * 256 + threadIdx.x;
    int e = t >> 5;            // 32 lanes per edge
    int lane = t & 31;
    if (e >= N_EDGES) return;
    int s = src[e], d = dst[e];
    const float4 v = *(const float4*)(x + (size_t)s * D_FEAT + lane * 4);
    float* ap = agg + (size_t)d * D_FEAT + lane * 4;
    unsafeAtomicAdd(ap + 0, v.x);
    unsafeAtomicAdd(ap + 1, v.y);
    unsafeAtomicAdd(ap + 2, v.z);
    unsafeAtomicAdd(ap + 3, v.w);
    if (lane == 0) unsafeAtomicAdd(deg + d, 1.0f);
}

// ---------------- kernel 2: nodes-per-graph count ----------------
__global__ __launch_bounds__(256) void count_nodes(const int* __restrict__ batch,
                                                   int* __restrict__ cnt) {
    int i = blockIdx.x * 256 + threadIdx.x;
    if (i < N_NODES) atomicAdd(cnt + batch[i], 1);
}

// ---------------- kernel 3: fused SAGE GEMMs + relu + pooling epilogue ----------------
// Each wave computes an M=32 x N=128 tile of h = relu(agg/deg @ Wl^T + x @ Wr^T + bl)
// and reduces it into per-graph max/sum pools. Two 16-row subtiles share every
// B fragment (4 WMMAs per LDS fragment pair); B fragments are double-buffered
// so LDS loads for n-tile nt+1 overlap the WMMA chain of n-tile nt.
__global__ __launch_bounds__(256) void sage_gemm_pool(const float* __restrict__ x,
                                                      const float* __restrict__ agg,
                                                      const float* __restrict__ deg,
                                                      const int* __restrict__ batch,
                                                      const float* __restrict__ Wl,
                                                      const float* __restrict__ bl,
                                                      const float* __restrict__ Wr,
                                                      int*   __restrict__ pmax,
                                                      float* __restrict__ psum) {
    __shared__ __bf16 sWl[HIDDEN * D_FEAT];   // [n][k] bf16, 32 KB
    __shared__ __bf16 sWr[HIDDEN * D_FEAT];   // 32 KB
    const int tid = threadIdx.x;
    for (int i = tid; i < HIDDEN * D_FEAT; i += 256) {
        sWl[i] = (__bf16)Wl[i];
        sWr[i] = (__bf16)Wr[i];
    }
    __syncthreads();

    const int lane  = tid & 31;
    const int wave  = tid >> 5;
    const int lmod  = lane & 15;
    const int lhalf = lane >> 4;

    for (int tile = blockIdx.x * 8 + wave; tile < N_TILES32; tile += gridDim.x * 8) {
        const int row0 = tile * 32;
        const int r0   = row0 + lmod;        // A row, subtile 0
        const int r1   = row0 + 16 + lmod;   // A row, subtile 1
        const float s0 = 1.0f / fmaxf(deg[r0], 1.0f);
        const float s1 = 1.0f / fmaxf(deg[r1], 1.0f);
        const float* aggRow0 = agg + (size_t)r0 * D_FEAT;
        const float* aggRow1 = agg + (size_t)r1 * D_FEAT;
        const float* xRow0   = x   + (size_t)r0 * D_FEAT;
        const float* xRow1   = x   + (size_t)r1 * D_FEAT;

        v8f acc0[8], acc1[8];
#pragma unroll
        for (int nt = 0; nt < 8; ++nt) { acc0[nt] = (v8f)0.0f; acc1[nt] = (v8f)0.0f; }

#pragma unroll
        for (int kt = 0; kt < 4; ++kt) {
            const int k0 = kt * 32 + lhalf * 8;
            v16bf aAgg0 = make_a_frag_f32(aggRow0, k0, s0);
            v16bf aX0   = make_a_frag_f32(xRow0,   k0, 1.0f);
            v16bf aAgg1 = make_a_frag_f32(aggRow1, k0, s1);
            v16bf aX1   = make_a_frag_f32(xRow1,   k0, 1.0f);

            v16bf bL[2], bR[2];
            bL[0] = make_frag_bf16(sWl + lmod * D_FEAT, k0);
            bR[0] = make_frag_bf16(sWr + lmod * D_FEAT, k0);
#pragma unroll
            for (int nt = 0; nt < 8; ++nt) {
                const int cur = nt & 1, nxt = cur ^ 1;
                if (nt < 7) {   // prefetch next n-tile's fragments into the other buffer
                    const int n = (nt + 1) * 16 + lmod;
                    bL[nxt] = make_frag_bf16(sWl + n * D_FEAT, k0);
                    bR[nxt] = make_frag_bf16(sWr + n * D_FEAT, k0);
                }
                acc0[nt] = wmma_bf16(aAgg0, bL[cur], acc0[nt]);
                acc1[nt] = wmma_bf16(aAgg1, bL[cur], acc1[nt]);
                acc0[nt] = wmma_bf16(aX0,   bR[cur], acc0[nt]);
                acc1[nt] = wmma_bf16(aX1,   bR[cur], acc1[nt]);
            }
        }

        // Epilogue: C/D layout -> M = lhalf*8 + r, N = lmod. relu then pool.
#pragma unroll
        for (int sub = 0; sub < 2; ++sub) {
#pragma unroll
            for (int r = 0; r < 8; ++r) {
                const int rowr = row0 + sub * 16 + lhalf * 8 + r;
                const int g = batch[rowr];
#pragma unroll
                for (int nt = 0; nt < 8; ++nt) {
                    const int n = nt * 16 + lmod;
                    float h = (sub == 0 ? acc0[nt][r] : acc1[nt][r]) + bl[n];
                    h = fmaxf(h, 0.0f);
                    // h >= 0 -> int-ordered max == float max; pools zero-initialized
                    atomicMax(pmax + g * HIDDEN + n, __float_as_int(h));
                    unsafeAtomicAdd(psum + g * HIDDEN + n, h);
                }
            }
        }
    }
}

// ---------------- kernel 4: build z = [max | mean] as bf16 ----------------
__global__ __launch_bounds__(256) void finalize_z(const int* __restrict__ pmax,
                                                  const float* __restrict__ psum,
                                                  const int* __restrict__ cnt,
                                                  __bf16* __restrict__ zbf) {
    int i = blockIdx.x * 256 + threadIdx.x;
    if (i >= N_GRAPHS * 2 * HIDDEN) return;
    int g = i >> 8;
    int c = i & 255;
    float v;
    if (c < HIDDEN) {
        v = __int_as_float(pmax[g * HIDDEN + c]);
    } else {
        float cn = (float)cnt[g];
        v = psum[g * HIDDEN + (c - HIDDEN)] / fmaxf(cn, 1.0f);
    }
    zbf[i] = (__bf16)v;
}

// ---------------- kernel 5: head GEMM [128x256]x[256x64] via WMMA ----------------
__global__ __launch_bounds__(256) void head_gemm(const __bf16* __restrict__ zbf,
                                                 const float* __restrict__ Wlin,
                                                 const float* __restrict__ blin,
                                                 float* __restrict__ out) {
    __shared__ __bf16 sW[FINAL_D * 2 * HIDDEN];   // [f][c] bf16, 32 KB
    const int tid = threadIdx.x;
    for (int i = tid; i < FINAL_D * 2 * HIDDEN; i += 256) sW[i] = (__bf16)Wlin[i];
    __syncthreads();

    const int lane  = tid & 31;
    const int mtile = tid >> 5;          // 8 waves -> 8 row tiles of 16 graphs
    const int lmod  = lane & 15;
    const int lhalf = lane >> 4;
    const int K2 = 2 * HIDDEN;

    v8f acc[4];
#pragma unroll
    for (int nt = 0; nt < 4; ++nt) acc[nt] = (v8f)0.0f;

#pragma unroll
    for (int kt = 0; kt < 8; ++kt) {
        const int k0 = kt * 32 + lhalf * 8;
        v16bf a = make_frag_bf16(zbf + (mtile * 16 + lmod) * K2, k0);
#pragma unroll
        for (int nt = 0; nt < 4; ++nt) {
            v16bf b = make_frag_bf16(sW + (nt * 16 + lmod) * K2, k0);
            acc[nt] = wmma_bf16(a, b, acc[nt]);
        }
    }

#pragma unroll
    for (int r = 0; r < 8; ++r) {
        const int g = mtile * 16 + lhalf * 8 + r;
#pragma unroll
        for (int nt = 0; nt < 4; ++nt) {
            const int f = nt * 16 + lmod;
            out[g * FINAL_D + f] = acc[nt][r] + blin[f];
        }
    }
}

extern "C" void kernel_launch(void* const* d_in, const int* in_sizes, int n_in,
                              void* d_out, int out_size, void* d_ws, size_t ws_size,
                              hipStream_t stream) {
    const float* x     = (const float*)d_in[0];
    const int*   eidx  = (const int*)d_in[1];     // [2, E] flat: src then dst
    const int*   batch = (const int*)d_in[2];
    const float* Wl    = (const float*)d_in[3];
    const float* bl    = (const float*)d_in[4];
    const float* Wr    = (const float*)d_in[5];
    const float* Wlin  = (const float*)d_in[6];
    const float* blin  = (const float*)d_in[7];
    float* out = (float*)d_out;

    char* ws = (char*)d_ws;
    float*  agg  = (float*)(ws + OFF_AGG);
    float*  deg  = (float*)(ws + OFF_DEG);
    int*    pmax = (int*)  (ws + OFF_PMAX);
    float*  psum = (float*)(ws + OFF_PSUM);
    int*    cnt  = (int*)  (ws + OFF_CNT);
    __bf16* zbf  = (__bf16*)(ws + OFF_ZBF);

    const int* src = eidx;
    const int* dst = eidx + N_EDGES;

    // 0) zero accumulators / pools / counters
    {
        size_t n16 = ZERO_BYTES / 16;
        int blocks = (int)((n16 + 255) / 256);
        if (blocks > 16384) blocks = 16384;
        init_ws<<<blocks, 256, 0, stream>>>((uint4*)ws, n16);
    }
    // 1) scatter-add neighbor features + degree
    {
        int blocks = (N_EDGES * 32 + 255) / 256;   // 80,000
        scatter_edges<<<blocks, 256, 0, stream>>>(x, src, dst, agg, deg);
    }
    // 2) per-graph node counts
    count_nodes<<<(N_NODES + 255) / 256, 256, 0, stream>>>(batch, cnt);
    // 3) fused SAGE GEMMs + relu + max/sum pooling (M=32 per wave)
    {
        int blocks = (N_TILES32 + 7) / 8;          // 391 workgroups x 8 waves
        sage_gemm_pool<<<blocks, 256, 0, stream>>>(x, agg, deg, batch,
                                                   Wl, bl, Wr, pmax, psum);
    }
    // 4) assemble z = [max | mean] in bf16
    finalize_z<<<(N_GRAPHS * 2 * HIDDEN + 255) / 256, 256, 0, stream>>>(pmax, psum, cnt, zbf);
    // 5) head linear via WMMA
    head_gemm<<<1, 256, 0, stream>>>(zbf, Wlin, blin, out);
}
